// BioDecoder_18769007084151
// MI455X (gfx1250) — matrix operand order repaired and checked by reading, not compiled, not run on hardware
//
#include <hip/hip_runtime.h>
#include <hip/hip_bf16.h>

// ---------------------------------------------------------------------------
// BioDecoder (LSTM caption decoder) for MI455X / gfx1250, wave32 + WMMA bf16.
//
// Dominant cost: (4088 x 256) @ (256 x 32000) output projection (~67 GFLOP,
// compute bound at 23.3 TB/s HBM) -> all GEMMs run on v_wmma_f32_16x16x32_bf16
// with f32 accumulation. The sequential LSTM cell math stays in f32 VALU.
//
// Recurrent kernel: w_hh is invariant across the 511-step time loop, so LICM
// tries to hoist ALL B-fragment loads out of the loop (256 VGPRs of weight
// data!) and spills them to scratch. We launder a zero byte-offset through an
// empty asm() each step: loads stay in-loop, but the base pointer keeps its
// global address-space provenance so they lower to global_load_b128 (not
// flat_load, which would also tie up DScnt).
// ---------------------------------------------------------------------------

typedef __bf16 bf16_t;
typedef __attribute__((ext_vector_type(16))) __bf16 v16bf;
typedef __attribute__((ext_vector_type(8)))  __bf16 v8bf;
typedef __attribute__((ext_vector_type(8)))  float  v8f;

#define VOCAB  32000
#define EMBED  128
#define HID    256
#define GATES  1024          // 4*HID
#define BATCH  8
#define TSTEPS 511           // T-1
#define MROWS  4088          // TSTEPS*BATCH
#define MPAD   4096          // padded to multiple of 16
#define MTILES (MPAD / 16)   // 256

// ---------------------------------------------------------------------------
// WMMA fragment loaders (layouts per CDNA5 ISA 7.12.2, 16-bit A 16x32 and
// B 32x16). We compute C = act @ W^T with W stored row-major (N x K), so a
// B-fragment "column" is a contiguous 32-element slice of a W row.
// ---------------------------------------------------------------------------
__device__ __forceinline__ v16bf load_a_frag(const bf16_t* Ab, int lda, int k0, int lane) {
    // Ab points at the 16-row tile base. lane&15 = M row; lane>>4 selects K-half.
    int m    = lane & 15;
    int half = lane >> 4;
    const bf16_t* p = Ab + (size_t)m * lda + k0 + half * 8;
    union { v16bf v; v8bf h[2]; } u;
    u.h[0] = *(const v8bf*)(p);          // K = k0 + half*8 .. +7
    u.h[1] = *(const v8bf*)(p + 16);     // K = k0 + 16 + half*8 .. +7
    return u.v;
}

__device__ __forceinline__ v16bf load_b_frag(const bf16_t* W, int ldw, int n0, int k0, int lane) {
    // lane&15 = N column -> row (n0+lane&15) of W; lane>>4 selects K-group of 8 (same as A).
    const bf16_t* p = W + (size_t)(n0 + (lane & 15)) * ldw + k0 + (lane >> 4) * 8;
    union { v16bf v; v8bf h[2]; } u;
    u.h[0] = *(const v8bf*)(p);
    u.h[1] = *(const v8bf*)(p + 16);
    return u.v;
}

__device__ __forceinline__ v8f wmma_bf16(v16bf a, v16bf b, v8f acc) {
    return __builtin_amdgcn_wmma_f32_16x16x32_bf16(
        /*neg_a=*/false, a, /*neg_b=*/false, b,
        /*c_mod=*/(short)0, acc, /*reuse_a=*/false, /*reuse_b=*/false);
}

__device__ __forceinline__ float sigf(float x) { return 1.0f / (1.0f + __expf(-x)); }

// ---------------------------------------------------------------------------
// Small utility kernels
// ---------------------------------------------------------------------------
__global__ void cvt_bf16_k(const float* __restrict__ in, bf16_t* __restrict__ out, int n) {
    int i = blockIdx.x * blockDim.x + threadIdx.x;
    if (i < n) out[i] = (bf16_t)in[i];
}

__global__ void zerorows_k(bf16_t* __restrict__ p, int startRow, int cols) {
    int i = blockIdx.x * blockDim.x + threadIdx.x;   // 8 rows * cols threads
    p[(size_t)startRow * cols + i] = (bf16_t)0.0f;
}

// Gather embeddings: row m = t*8 + b  ->  emb_w[captions[b][t]], pad rows zero.
__global__ void embed_k(const int* __restrict__ captions,
                        const float* __restrict__ emb,
                        bf16_t* __restrict__ xbf) {
    int m = blockIdx.x;          // 0..MPAD-1
    int k = threadIdx.x;         // 0..EMBED-1
    bf16_t v = (bf16_t)0.0f;
    if (m < MROWS) {
        int t = m >> 3, b = m & 7;
        int tok = captions[b * 512 + t];
        v = (bf16_t)emb[(size_t)tok * EMBED + k];
    }
    xbf[(size_t)m * EMBED + k] = v;
}

// ---------------------------------------------------------------------------
// Generic bf16 WMMA GEMM:  C(M,N) = A(M,K) @ W(N,K)^T + bias0 + bias1
// One 16x16 tile per wave, NREP column tiles share the A fragments.
// Block = 256 threads = 8 waves.  grid.x * 8 == MTILES * (N/16/NREP).
// Tiles are scheduled M-fastest so concurrent waves share the same W slice
// (better L2 locality for the 16MB out_w stream; A is fully L2 resident).
// ---------------------------------------------------------------------------
template <int KTILES, int NREP, int ACT>
__global__ __launch_bounds__(256) void gemm_bf16_k(
    const bf16_t* __restrict__ A, const bf16_t* __restrict__ W,
    const float* __restrict__ bias0, const float* __restrict__ bias1,
    float* __restrict__ Cf, bf16_t* __restrict__ Cb,
    int Mvalid, int NTG, int ldc)
{
    const int K    = KTILES * 32;
    const int wave = threadIdx.x >> 5;
    const int lane = threadIdx.x & 31;
    const int tile = blockIdx.x * 8 + wave;
    const int mt   = tile % MTILES;        // M fastest
    const int ng   = tile / MTILES;
    const int m0   = mt * 16;
    (void)NTG;

    v16bf afrag[KTILES];
#pragma unroll
    for (int kf = 0; kf < KTILES; ++kf)
        afrag[kf] = load_a_frag(A + (size_t)m0 * K, K, kf * 32, lane);

#pragma unroll
    for (int j = 0; j < NREP; ++j) {
        const int n0 = (ng * NREP + j) * 16;
        v8f acc;
#pragma unroll
        for (int r = 0; r < 8; ++r) acc[r] = 0.0f;
#pragma unroll
        for (int kf = 0; kf < KTILES; ++kf) {
            v16bf b = load_b_frag(W, K, n0, kf * 32, lane);
            acc = wmma_bf16(afrag[kf], b, acc);
        }
        const int n = n0 + (lane & 15);
        float bias = (bias0 ? bias0[n] : 0.0f) + (bias1 ? bias1[n] : 0.0f);
#pragma unroll
        for (int r = 0; r < 8; ++r) {
            int m = m0 + (lane >> 4) * 8 + r;
            float v = acc[r] + bias;
            if (ACT == 1) v = tanhf(v);
            if (m < Mvalid) {
                if (Cf) Cf[(size_t)m * ldc + n] = v;
                if (Cb) Cb[(size_t)m * ldc + n] = (bf16_t)v;
            }
        }
    }
}

// ---------------------------------------------------------------------------
// Sequential LSTM layer. One workgroup, 512 threads = 16 waves (4 waves/SIMD
// for latency hiding; ~150 VGPRs/wave fits the budget now that the weight
// loads are not hoisted). Per step: gates(8,1024) = xp[t] + h @ w_hh^T via
// WMMA with the accumulator preloaded from xp; each wave owns 4 N-tiles and
// reuses the 8 A-fragments (h in LDS as a 16x256 bf16 tile, rows 8..15 zero)
// across them. Then f32 sigmoid/tanh cell update in LDS.
// LDS: 8KB h + 8KB c + 32KB gates = 48KB.
// ---------------------------------------------------------------------------
__global__ __launch_bounds__(512) void lstm_seq_k(
    const float*  __restrict__ xp,        // (MPAD, 1024) rows t*8+b
    const bf16_t* __restrict__ whh,       // (1024, 256) bf16 row-major
    const float*  __restrict__ h0,        // (8, 256)
    bf16_t* __restrict__ h_out_bf,        // (MPAD, 256)
    float*  __restrict__ h_out_f32)       // nullable (MPAD, 256)
{
    __shared__ bf16_t h_sh[16 * HID];
    __shared__ float  c_sh[BATCH * HID];
    __shared__ float  g_sh[BATCH * GATES];

    const int tid  = threadIdx.x;
    const int wave = tid >> 5;            // 0..15
    const int lane = tid & 31;

    for (int i = tid; i < 16 * HID; i += 512) {
        int m = i / HID, j = i % HID;
        h_sh[i] = (m < BATCH) ? (bf16_t)h0[m * HID + j] : (bf16_t)0.0f;
    }
    for (int i = tid; i < BATCH * HID; i += 512) c_sh[i] = 0.0f;
    __syncthreads();

    size_t zoff = 0;   // always 0, but the compiler must not assume so
    for (int t = 0; t < TSTEPS; ++t) {
        // Launder a zero element-offset: w_hh loads cannot be hoisted across
        // steps, yet the base pointer keeps its global provenance (global_load).
        asm volatile("" : "+s"(zoff));
        const bf16_t* whh_t = whh + zoff;

        // ---- A fragments: h (16x256 bf16) from LDS, loaded once per step ----
        v16bf afrag[8];
#pragma unroll
        for (int kf = 0; kf < 8; ++kf)
            afrag[kf] = load_a_frag(h_sh, HID, kf * 32, lane);

        // ---- gates = xp[t] + h @ whh^T  (M=16 padded, N=1024, K=256) ----
#pragma unroll
        for (int w = 0; w < 4; ++w) {
            const int nt = wave * 4 + w;
            const int n0 = nt * 16;
            const int n  = n0 + (lane & 15);
            v8f acc;
            if (lane < 16) {
                const float* xr = xp + (size_t)(t * BATCH) * GATES + n;
#pragma unroll
                for (int r = 0; r < 8; ++r) acc[r] = xr[(size_t)r * GATES];
            } else {
#pragma unroll
                for (int r = 0; r < 8; ++r) acc[r] = 0.0f;
            }
#pragma unroll
            for (int kf = 0; kf < 8; ++kf) {
                v16bf b = load_b_frag(whh_t, HID, n0, kf * 32, lane);
                acc = wmma_bf16(afrag[kf], b, acc);
            }
            if (lane < 16) {   // rows 0..7 live in lanes 0..15
#pragma unroll
                for (int r = 0; r < 8; ++r) g_sh[r * GATES + n] = acc[r];
            }
        }
        __syncthreads();

        // ---- f32 cell update: 2048 (b, j) elements over 512 threads ----
#pragma unroll
        for (int e = 0; e < 4; ++e) {
            const int idx = tid + e * 512;
            const int b = idx >> 8, j = idx & 255;
            float gi = g_sh[b * GATES + j];
            float gf = g_sh[b * GATES + 256 + j];
            float gg = g_sh[b * GATES + 512 + j];
            float go = g_sh[b * GATES + 768 + j];
            float c  = c_sh[idx];
            c = sigf(gf) * c + sigf(gi) * tanhf(gg);
            float h = sigf(go) * tanhf(c);
            c_sh[idx] = c;
            bf16_t hb = (bf16_t)h;
            h_sh[b * HID + j] = hb;
            size_t o = (size_t)(t * BATCH + b) * HID + j;
            h_out_bf[o] = hb;
            if (h_out_f32) h_out_f32[o] = h;
        }
        __syncthreads();
    }
}

// ---------------------------------------------------------------------------
// Gate head: sigz[m] = sigmoid( dot(g1[m], gate_w2) + gate_b2 ), wave per row.
// ---------------------------------------------------------------------------
__global__ __launch_bounds__(256) void gate_dot_k(
    const float* __restrict__ g1, const float* __restrict__ w2,
    const float* __restrict__ b2, float* __restrict__ sigz, int M)
{
    const int wave = threadIdx.x >> 5;
    const int lane = threadIdx.x & 31;
    const int m = blockIdx.x * 8 + wave;
    if (m >= M) return;
    const float* row = g1 + (size_t)m * HID;
    float s = 0.0f;
    for (int j = lane; j < HID; j += 32) s += row[j] * w2[j];
#pragma unroll
    for (int off = 16; off > 0; off >>= 1) s += __shfl_down(s, off, 32);
    if (lane == 0) sigz[m] = sigf(s + b2[0]);
}

// gated[m][k] = bf16( h2[m][k] * sigz[m] ), pad rows zero.
__global__ void gated_k(const float* __restrict__ h2, const float* __restrict__ sigz,
                        bf16_t* __restrict__ out)
{
    const int idx = blockIdx.x * blockDim.x + threadIdx.x;   // MPAD*HID
    const int m = idx >> 8;
    float v = (m < MROWS) ? h2[idx] * sigz[m] : 0.0f;
    out[idx] = (bf16_t)v;
}

// ---------------------------------------------------------------------------
// Launch
// ---------------------------------------------------------------------------
extern "C" void kernel_launch(void* const* d_in, const int* in_sizes, int n_in,
                              void* d_out, int out_size, void* d_ws, size_t ws_size,
                              hipStream_t stream) {
    const float* thought  = (const float*)d_in[0];
    const int*   captions = (const int*)  d_in[1];
    const float* emb_w    = (const float*)d_in[2];
    const float* w_ih_l0  = (const float*)d_in[3];
    const float* w_hh_l0  = (const float*)d_in[4];
    const float* b_ih_l0  = (const float*)d_in[5];
    const float* b_hh_l0  = (const float*)d_in[6];
    const float* w_ih_l1  = (const float*)d_in[7];
    const float* w_hh_l1  = (const float*)d_in[8];
    const float* b_ih_l1  = (const float*)d_in[9];
    const float* b_hh_l1  = (const float*)d_in[10];
    const float* gate_w1  = (const float*)d_in[11];
    const float* gate_b1  = (const float*)d_in[12];
    const float* gate_w2  = (const float*)d_in[13];
    const float* gate_b2  = (const float*)d_in[14];
    const float* out_w    = (const float*)d_in[15];
    const float* out_b    = (const float*)d_in[16];
    float* out = (float*)d_out;

    char* base = (char*)d_ws;
    size_t off = 0;
    auto alloc = [&](size_t bytes) -> char* {
        char* p = base + off;
        off += (bytes + 255) & ~(size_t)255;
        return p;
    };
    bf16_t* x_bf     = (bf16_t*)alloc((size_t)MPAD * EMBED * 2);
    bf16_t* wih0_bf  = (bf16_t*)alloc((size_t)GATES * EMBED * 2);
    bf16_t* whh0_bf  = (bf16_t*)alloc((size_t)GATES * HID * 2);
    bf16_t* wih1_bf  = (bf16_t*)alloc((size_t)GATES * HID * 2);
    bf16_t* whh1_bf  = (bf16_t*)alloc((size_t)GATES * HID * 2);
    bf16_t* gw1_bf   = (bf16_t*)alloc((size_t)HID * HID * 2);
    bf16_t* outw_bf  = (bf16_t*)alloc((size_t)VOCAB * HID * 2);
    float*  xp0      = (float*) alloc((size_t)MPAD * GATES * 4);
    float*  xp1      = (float*) alloc((size_t)MPAD * GATES * 4);
    bf16_t* h1_bf    = (bf16_t*)alloc((size_t)MPAD * HID * 2);
    bf16_t* h2_bf    = (bf16_t*)alloc((size_t)MPAD * HID * 2);
    float*  h2_f     = (float*) alloc((size_t)MPAD * HID * 4);
    float*  g1_f     = (float*) alloc((size_t)MPAD * HID * 4);
    float*  sigz     = (float*) alloc((size_t)MPAD * 4);
    bf16_t* gated_bf = (bf16_t*)alloc((size_t)MPAD * HID * 2);
    (void)ws_size; (void)n_in; (void)in_sizes; (void)out_size;

    // --- weight conversions to bf16 ---
    auto cvt = [&](const float* src, bf16_t* dst, int n) {
        cvt_bf16_k<<<(n + 255) / 256, 256, 0, stream>>>(src, dst, n);
    };
    cvt(w_ih_l0, wih0_bf, GATES * EMBED);
    cvt(w_hh_l0, whh0_bf, GATES * HID);
    cvt(w_ih_l1, wih1_bf, GATES * HID);
    cvt(w_hh_l1, whh1_bf, GATES * HID);
    cvt(gate_w1, gw1_bf,  HID * HID);
    cvt(out_w,   outw_bf, VOCAB * HID);

    // --- embedding gather (zero-padded to MPAD rows) ---
    embed_k<<<MPAD, EMBED, 0, stream>>>(captions, emb_w, x_bf);

    // clear the pad rows of LSTM output buffers (they feed later GEMM A tiles)
    zerorows_k<<<8, 256, 0, stream>>>(h1_bf, MROWS, HID);
    zerorows_k<<<8, 256, 0, stream>>>(h2_bf, MROWS, HID);

    // --- xp0 = x @ w_ih_l0^T + b_ih_l0 + b_hh_l0   (M=4096, N=1024, K=128)
    gemm_bf16_k<4, 2, 0><<<MTILES * (GATES / 32) / 8, 256, 0, stream>>>(
        x_bf, wih0_bf, b_ih_l0, b_hh_l0, xp0, nullptr, MPAD, GATES / 32, GATES);

    // --- LSTM layer 0 (sequential) ---
    lstm_seq_k<<<1, 512, 0, stream>>>(xp0, whh0_bf, thought, h1_bf, nullptr);

    // --- xp1 = h1 @ w_ih_l1^T + b_ih_l1 + b_hh_l1  (M=4096, N=1024, K=256)
    gemm_bf16_k<8, 2, 0><<<MTILES * (GATES / 32) / 8, 256, 0, stream>>>(
        h1_bf, wih1_bf, b_ih_l1, b_hh_l1, xp1, nullptr, MPAD, GATES / 32, GATES);

    // --- LSTM layer 1 (sequential) ---
    lstm_seq_k<<<1, 512, 0, stream>>>(xp1, whh1_bf, thought, h2_bf, h2_f);

    // --- gate hidden: g1 = tanh(h2 @ gate_w1^T + gate_b1)  (N=256, K=256)
    gemm_bf16_k<8, 2, 1><<<MTILES * (HID / 32) / 8, 256, 0, stream>>>(
        h2_bf, gw1_bf, gate_b1, nullptr, g1_f, nullptr, MPAD, HID / 32, HID);

    // --- sigz[m] = sigmoid(g1[m] . gate_w2 + gate_b2) ---
    gate_dot_k<<<MPAD / 8, 256, 0, stream>>>(g1_f, gate_w2, gate_b2, sigz, MPAD);

    // --- gated = h2 * sigz (bf16, pad rows zero) ---
    gated_k<<<(MPAD * HID) / 256, 256, 0, stream>>>(h2_f, sigz, gated_bf);

    // --- output projection: out = gated @ out_w^T + out_b
    //     (M=4088 valid of 4096, N=32000, K=256), 16x64 strip per wave.
    gemm_bf16_k<8, 4, 0><<<MTILES * (VOCAB / 64) / 8, 256, 0, stream>>>(
        gated_bf, outw_bf, out_b, nullptr, out, nullptr, MROWS, VOCAB / 64, VOCAB);
}
